// HypER_16552803959284
// MI455X (gfx1250) — matrix operand order repaired and checked by reading, not compiled, not run on hardware
//
#include <hip/hip_runtime.h>
#include <math.h>

typedef __attribute__((ext_vector_type(16))) _Float16 v16h;
typedef __attribute__((ext_vector_type(8)))  _Float16 v8h;
typedef __attribute__((ext_vector_type(8)))  float    v8f;
typedef __attribute__((ext_vector_type(4)))  float    v4f;

#define NE_     128000
#define D1_     200
#define D2_     200
#define BATCH_  1024
#define OUT_C_  32
#define FW_     9
#define LCONV_  192      // D1 - FW + 1
#define FC_LEN_ 6144     // OUT_C * LCONV
#define FC1_LEN_ 288     // OUT_C * FW
#define KPAD_   224      // D1 padded up to multiple of 32
#define ESTRIDE_ 232     // LDS f16 row stride (mult of 8 -> 16B aligned, conflict-free)
#define EPS_    1e-5f

// ---------------------------------------------------------------------------
// Kernel 1: k = R[r_idx] @ fc1_w.T + fc1_b ; x = BN0(E[e1_idx]) ; zero h16 pad
// ---------------------------------------------------------------------------
__global__ void __launch_bounds__(256)
prep_kernel(const float* __restrict__ R, const float* __restrict__ fc1_w,
            const float* __restrict__ fc1_b, const float* __restrict__ E,
            const float* __restrict__ g0, const float* __restrict__ b0,
            const float* __restrict__ m0, const float* __restrict__ v0,
            const int* __restrict__ e1_idx, const int* __restrict__ r_idx,
            float* __restrict__ kbuf, float* __restrict__ xbuf,
            _Float16* __restrict__ h16)
{
    __shared__ float rs[D2_];
    const int b = blockIdx.x;
    const int t = threadIdx.x;

    const int ri = r_idx[b];
    for (int d = t; d < D2_; d += blockDim.x) rs[d] = R[(size_t)ri * D2_ + d];
    __syncthreads();

    for (int j = t; j < FC1_LEN_; j += blockDim.x) {
        const float* w = fc1_w + (size_t)j * D2_;
        float s = fc1_b[j];
        for (int d = 0; d < D2_; ++d) s = fmaf(rs[d], w[d], s);
        kbuf[(size_t)b * FC1_LEN_ + j] = s;
    }

    const int ei = e1_idx[b];
    const float scale0 = g0[0] * rsqrtf(v0[0] + EPS_);
    const float mean0  = m0[0];
    const float beta0  = b0[0];
    for (int d = t; d < D1_; d += blockDim.x)
        xbuf[(size_t)b * D1_ + d] = (E[(size_t)ei * D1_ + d] - mean0) * scale0 + beta0;

    // zero the K padding of h16 rows (cols 200..223)
    for (int n = D1_ + t; n < KPAD_; n += blockDim.x)
        h16[(size_t)b * KPAD_ + n] = (_Float16)0.0f;
}

// ---------------------------------------------------------------------------
// Kernel 2: conv[b][o][l] = sum_f x[b][l+f]*k[b][o][f], BN1, store as f16
// ---------------------------------------------------------------------------
__global__ void __launch_bounds__(256)
conv_kernel(const float* __restrict__ kbuf, const float* __restrict__ xbuf,
            const float* __restrict__ g1, const float* __restrict__ b1,
            const float* __restrict__ m1, const float* __restrict__ v1,
            _Float16* __restrict__ conv16)
{
    __shared__ float xs[D1_];
    __shared__ float ks[FC1_LEN_];
    __shared__ float sc[OUT_C_];
    __shared__ float of[OUT_C_];
    const int b = blockIdx.x;
    const int t = threadIdx.x;

    for (int d = t; d < D1_; d += blockDim.x)      xs[d] = xbuf[(size_t)b * D1_ + d];
    for (int j = t; j < FC1_LEN_; j += blockDim.x) ks[j] = kbuf[(size_t)b * FC1_LEN_ + j];
    if (t < OUT_C_) {
        float s = g1[t] * rsqrtf(v1[t] + EPS_);
        sc[t] = s;
        of[t] = b1[t] - m1[t] * s;
    }
    __syncthreads();

    for (int idx = t; idx < FC_LEN_; idx += blockDim.x) {
        const int o = idx / LCONV_;
        const int l = idx % LCONV_;
        float s = 0.0f;
#pragma unroll
        for (int f = 0; f < FW_; ++f) s = fmaf(xs[l + f], ks[o * FW_ + f], s);
        conv16[(size_t)b * FC_LEN_ + idx] = (_Float16)(s * sc[o] + of[o]);
    }
}

// ---------------------------------------------------------------------------
// Kernel 3: h = conv16(1024x6144) @ fc_w.T(6144x200) + fc_b ; BN2 ; ReLU -> h16
// One wave per 16x16 output tile. grid = (13 N-tiles, 8 M-strips), 8 waves/block
// ---------------------------------------------------------------------------
__global__ void __launch_bounds__(256)
fc_wmma_kernel(const _Float16* __restrict__ conv16, const float* __restrict__ fc_w,
               const float* __restrict__ fc_b,
               const float* __restrict__ g2, const float* __restrict__ b2,
               const float* __restrict__ m2, const float* __restrict__ v2,
               _Float16* __restrict__ h16)
{
    const int lane  = threadIdx.x & 31;
    const int wave  = threadIdx.x >> 5;
    const int ntile = blockIdx.x;              // 0..12 (200 cols -> 13 tiles)
    const int mtile = blockIdx.y * 8 + wave;   // 0..63
    const int mloc  = lane & 15;
    const int hi    = lane >> 4;               // K-half selector
    const int n     = ntile * 16 + mloc;       // output column (== B fragment col)
    const int mrow  = mtile * 16 + mloc;       // A fragment row

    v8f acc = {};
    const _Float16* arow = conv16 + (size_t)mrow * FC_LEN_;
    const bool nvalid = (n < D1_);
    const float* brow = fc_w + (size_t)(nvalid ? n : 0) * FC_LEN_;

    for (int k = 0; k < FC_LEN_; k += 32) {
        // A: 16x32 f16 fragment — two contiguous 8-halfword chunks per lane
        v16h a;
        const int ka = k + hi * 16;
        *(v8h*)&a       = *(const v8h*)(arow + ka);
        *((v8h*)&a + 1) = *(const v8h*)(arow + ka + 8);

        // B: 32x16 f16 fragment — 16 contiguous K per lane, f32 -> f16 convert
        v16h bb = {};
        if (nvalid) {
            const int kb = k + hi * 16;
            v4f f0 = *(const v4f*)(brow + kb);
            v4f f1 = *(const v4f*)(brow + kb + 4);
            v4f f2 = *(const v4f*)(brow + kb + 8);
            v4f f3 = *(const v4f*)(brow + kb + 12);
#pragma unroll
            for (int i = 0; i < 4; ++i) {
                bb[i]      = (_Float16)f0[i];
                bb[4 + i]  = (_Float16)f1[i];
                bb[8 + i]  = (_Float16)f2[i];
                bb[12 + i] = (_Float16)f3[i];
            }
        }
        acc = __builtin_amdgcn_wmma_f32_16x16x32_f16(false, a, false, bb,
                                                     (short)0, acc, false, false);
    }

    if (nvalid) {
        const float scale = g2[n] * rsqrtf(v2[n] + EPS_);
        const float mean  = m2[n];
        const float beta  = b2[n];
        const float bias  = fc_b[n];
#pragma unroll
        for (int j = 0; j < 8; ++j) {
            const int m = mtile * 16 + j + 8 * hi;
            float hv = acc[j] + bias;
            hv = (hv - mean) * scale + beta;
            hv = fmaxf(hv, 0.0f);
            h16[(size_t)m * KPAD_ + n] = (_Float16)hv;
        }
    }
}

// ---------------------------------------------------------------------------
// Kernel 4: out = sigmoid(h16(1024x224) @ E.T + b). Dominant kernel.
//
// Block: 8 waves, covers 512 rows x 64 entity cols. The block's E column tile
// (64 rows x 224 K) is converted f32->f16 once into LDS; each wave then
// computes a 64x64 output tile (4 M-tiles x 4 N-tiles) so every A fragment
// (global, L2-resident h16) is reused 4x and every B fragment (ds_load_b128
// from LDS) is reused 4x. 16 WMMAs per k-step, 7 k-steps (K padded 200->224).
// grid = (2000 N-groups, 2 M-halves).
// ---------------------------------------------------------------------------
__global__ void __launch_bounds__(256)
logits_kernel(const _Float16* __restrict__ h16, const float* __restrict__ E,
              const float* __restrict__ bias, float* __restrict__ out)
{
    __shared__ _Float16 etile[64][ESTRIDE_];   // 29,696 B

    const int lane  = threadIdx.x & 31;
    const int wave  = threadIdx.x >> 5;
    const int t     = threadIdx.x;
    const int n0    = blockIdx.x * 64;               // first entity column
    const int mbase = blockIdx.y * 512 + wave * 64;  // first row of wave tile
    const int mloc  = lane & 15;
    const int hi    = lane >> 4;

    // Cooperative fill: E[n0..n0+63][0..199] -> f16, zero-padded to ESTRIDE_.
    for (int idx = t; idx < 64 * ESTRIDE_; idx += 256) {
        const int row = idx / ESTRIDE_;
        const int c   = idx % ESTRIDE_;
        float v = 0.0f;
        if (c < D1_) v = E[(size_t)(n0 + row) * D1_ + c];
        etile[row][c] = (_Float16)v;
    }

    __builtin_prefetch(h16 + (size_t)(mbase + mloc) * KPAD_, 0, 0);
    __syncthreads();

    v8f acc[4][4] = {};

#pragma unroll
    for (int kstep = 0; kstep < 7; ++kstep) {
        const int k  = kstep * 32;
        const int ka = k + hi * 16;
        const int kb = k + hi * 16;

        // B fragments from LDS (two ds_load_b128 each)
        v16h bb[4];
#pragma unroll
        for (int nt = 0; nt < 4; ++nt) {
            const _Float16* er = &etile[nt * 16 + mloc][0];
            *(v8h*)&bb[nt]       = *(const v8h*)(er + kb);
            *((v8h*)&bb[nt] + 1) = *(const v8h*)(er + kb + 8);
        }

        // A fragments from global h16 (L2-resident)
        v16h a[4];
#pragma unroll
        for (int mt = 0; mt < 4; ++mt) {
            const _Float16* arow =
                h16 + (size_t)(mbase + mt * 16 + mloc) * KPAD_;
            *(v8h*)&a[mt]       = *(const v8h*)(arow + ka);
            *((v8h*)&a[mt] + 1) = *(const v8h*)(arow + ka + 8);
        }

#pragma unroll
        for (int mt = 0; mt < 4; ++mt)
#pragma unroll
            for (int nt = 0; nt < 4; ++nt)
                acc[mt][nt] = __builtin_amdgcn_wmma_f32_16x16x32_f16(
                    false, a[mt], false, bb[nt], (short)0, acc[mt][nt],
                    false, false);
    }

    // Epilogue: + entity bias, sigmoid, store
#pragma unroll
    for (int nt = 0; nt < 4; ++nt) {
        const int n = n0 + nt * 16 + mloc;
        const float bn = bias[n];
#pragma unroll
        for (int mt = 0; mt < 4; ++mt) {
#pragma unroll
            for (int j = 0; j < 8; ++j) {
                const int m = mbase + mt * 16 + j + 8 * hi;
                const float v = acc[mt][nt][j] + bn;
                out[(size_t)m * NE_ + n] = 1.0f / (1.0f + __expf(-v));
            }
        }
    }
}

// ---------------------------------------------------------------------------
extern "C" void kernel_launch(void* const* d_in, const int* in_sizes, int n_in,
                              void* d_out, int out_size, void* d_ws, size_t ws_size,
                              hipStream_t stream)
{
    const float* E      = (const float*)d_in[0];
    const float* R      = (const float*)d_in[1];
    const float* fc1_w  = (const float*)d_in[2];
    const float* fc1_b  = (const float*)d_in[3];
    const float* fc_w   = (const float*)d_in[4];
    const float* fc_b   = (const float*)d_in[5];
    const float* bn0_g  = (const float*)d_in[6];
    const float* bn0_b  = (const float*)d_in[7];
    const float* bn0_m  = (const float*)d_in[8];
    const float* bn0_v  = (const float*)d_in[9];
    const float* bn1_g  = (const float*)d_in[10];
    const float* bn1_b  = (const float*)d_in[11];
    const float* bn1_m  = (const float*)d_in[12];
    const float* bn1_v  = (const float*)d_in[13];
    const float* bn2_g  = (const float*)d_in[14];
    const float* bn2_b  = (const float*)d_in[15];
    const float* bn2_m  = (const float*)d_in[16];
    const float* bn2_v  = (const float*)d_in[17];
    const float* bvec   = (const float*)d_in[18];
    const int*   e1_idx = (const int*)d_in[19];
    const int*   r_idx  = (const int*)d_in[20];
    float* out = (float*)d_out;

    // workspace layout (256B-aligned slabs, ~15 MB total)
    char* ws = (char*)d_ws;
    float*     kbuf   = (float*)(ws + 0);                       //  1024*288*4  = 1,179,648
    float*     xbuf   = (float*)(ws + 1179648);                 //  1024*200*4  =   819,200
    _Float16*  conv16 = (_Float16*)(ws + 1998848);              //  1024*6144*2 = 12,582,912
    _Float16*  h16    = (_Float16*)(ws + 14581760);             //  1024*224*2  =   458,752

    prep_kernel<<<BATCH_, 256, 0, stream>>>(R, fc1_w, fc1_b, E,
                                            bn0_g, bn0_b, bn0_m, bn0_v,
                                            e1_idx, r_idx, kbuf, xbuf, h16);

    conv_kernel<<<BATCH_, 256, 0, stream>>>(kbuf, xbuf,
                                            bn1_g, bn1_b, bn1_m, bn1_v, conv16);

    fc_wmma_kernel<<<dim3(13, 8), 256, 0, stream>>>(conv16, fc_w, fc_b,
                                                    bn2_g, bn2_b, bn2_m, bn2_v, h16);

    logits_kernel<<<dim3(NE_ / 64, 2), 256, 0, stream>>>(h16, E, bvec, out);
}
